// DownSampleBlock_7919919693899
// MI455X (gfx1250) — compile-verified, matrix operand order checked by reading, not run on hardware
//
#include <hip/hip_runtime.h>
#include <hip/hip_bf16.h>
#include <math.h>

typedef __attribute__((ext_vector_type(16))) _Float16 v16h;
typedef __attribute__((ext_vector_type(8)))  float    v8f;

// Map float -> uint such that larger float => larger uint (total order).
__device__ __forceinline__ unsigned int orderKey(float s) {
    unsigned int u = __float_as_uint(s);
    return (u & 0x80000000u) ? ~u : (u | 0x80000000u);
}

// ---------------------------------------------------------------------------
// Stage 1: scores via WMMA. Each wave computes 16 node scores with one
// v_wmma_f32_16x16x32_f16:  A = 16 rows of x (K=0..F-1, rest 0),
// B[k][n] = w[k] for all n.  D column 0 (lanes 0 and 16) holds the 16 dots.
// ---------------------------------------------------------------------------
__global__ void score_wmma_kernel(const float* __restrict__ x,
                                  const float* __restrict__ w,
                                  int N, int F,
                                  unsigned int* __restrict__ keys,
                                  float* __restrict__ scores) {
    const int lane = threadIdx.x & 31;
    const int wave = (int)((blockIdx.x * blockDim.x + threadIdx.x) >> 5);
    const int base = wave * 16;
    if (base >= N) return;                      // wave-uniform branch; EXEC stays full

    float nw = 0.f;
    for (int f = 0; f < F; ++f) { float wf = w[f]; nw += wf * wf; }
    const float inv_norm = rsqrtf(nw);

    // A (16x32 f16): lanes 0-15 hold row M=lane; halfs a[0..7] are K=0..7.
    // Lanes 16-31 correspond to K=8..15 etc -> zero (F<=8).
    v16h a = {};
    v16h b = {};
    if (lane < 16) {
        const int m = base + lane;
        if (m < N) {
            for (int f = 0; f < F && f < 8; ++f)
                a[f] = (_Float16)x[(long long)m * F + f];
        }
        // B (32x16 f16): lanes 0-15 hold rows K=0..7 and K=16..23 (2 per VGPR); col = lane.
        for (int f = 0; f < F && f < 8; ++f)
            b[f] = (_Float16)w[f];
    }
    v8f c = {};
    c = __builtin_amdgcn_wmma_f32_16x16x32_f16(false, a, false, b,
                                               (short)0, c, false, false);
    // D layout: VGPR v, lanes 0-15 -> (M=v, N=lane); lanes 16-31 -> (M=v+8).
    // Column N=0 lives in lane 0 (M=0..7) and lane 16 (M=8..15).
    if (lane == 0 || lane == 16) {
        const int moff = (lane == 0) ? 0 : 8;
        for (int v = 0; v < 8; ++v) {
            const int m = base + moff + v;
            if (m < N) {
                const float s = c[v] * inv_norm;
                scores[m] = s;
                keys[m]   = orderKey(s);
            }
        }
    }
}

// ---------------------------------------------------------------------------
// Stage 2: exact radix-select of the k-th largest key (MSB-first, 4x8 bits).
// state[0]=prefix, state[1]=remaining, state[2]=cnt_gt, state[3]=cnt_eq.
// ---------------------------------------------------------------------------
__global__ void init_kernel(unsigned int* __restrict__ hist4,
                            unsigned int* __restrict__ state,
                            unsigned int k) {
    const int t = (int)(blockIdx.x * blockDim.x + threadIdx.x);
    if (t < 1024) hist4[t] = 0u;
    if (t == 0) { state[0] = 0u; state[1] = k; state[2] = 0u; state[3] = 0u; }
}

__global__ void hist_kernel(const unsigned int* __restrict__ keys, int N,
                            const unsigned int* __restrict__ state,
                            unsigned int* __restrict__ hist, int pass) {
    __shared__ unsigned int lh[256];
    for (int t = threadIdx.x; t < 256; t += blockDim.x) lh[t] = 0u;
    __syncthreads();
    const int shift = 24 - 8 * pass;
    const unsigned int prefix = state[0];
    const unsigned int himask = (pass == 0) ? 0u : (0xFFFFFFFFu << (shift + 8));
    int i = (int)(blockIdx.x * blockDim.x + threadIdx.x);
    const int stride = (int)(gridDim.x * blockDim.x);
    for (; i < N; i += stride) {
        const unsigned int kk = keys[i];
        if (((kk ^ prefix) & himask) == 0u)
            atomicAdd(&lh[(kk >> shift) & 255u], 1u);
    }
    __syncthreads();
    for (int t = threadIdx.x; t < 256; t += blockDim.x)
        if (lh[t]) atomicAdd(&hist[t], lh[t]);
}

__global__ void scan_kernel(const unsigned int* __restrict__ hist,
                            unsigned int* __restrict__ state, int pass) {
    if (threadIdx.x != 0) return;               // 256 serial iters, negligible
    const int shift = 24 - 8 * pass;
    const unsigned int remaining = state[1];
    unsigned int acc = 0u;
    for (int b = 255; b >= 0; --b) {
        const unsigned int h = hist[b];
        if (acc + h >= remaining) {
            state[0] |= ((unsigned int)b) << shift;
            state[1]  = remaining - acc;        // how many == entries still needed
            return;
        }
        acc += h;
    }
}

__global__ void compact_kernel(const unsigned int* __restrict__ keys, int N,
                               unsigned int* __restrict__ state,
                               int* __restrict__ nidx,
                               unsigned int* __restrict__ perm,
                               unsigned int k) {
    const unsigned int T       = state[0];
    const unsigned int eq_take = state[1];
    const unsigned int k_gt    = k - eq_take;
    int i = (int)(blockIdx.x * blockDim.x + threadIdx.x);
    const int stride = (int)(gridDim.x * blockDim.x);
    for (; i < N; i += stride) {
        const unsigned int kk = keys[i];
        int pos = -1;
        if (kk > T) {
            pos = (int)atomicAdd(&state[2], 1u);
        } else if (kk == T) {
            const unsigned int t = atomicAdd(&state[3], 1u);
            if (t < eq_take) pos = (int)(k_gt + t);
        }
        nidx[i] = pos;                          // int32 remap table: 4 MB, L2-resident
        if (pos >= 0) perm[pos] = (unsigned int)i;
    }
}

// ---------------------------------------------------------------------------
// Stage 3: x_out[row] = x[perm[row]] * tanh(score)
// ---------------------------------------------------------------------------
__global__ void xout_kernel(const float* __restrict__ x,
                            const unsigned int* __restrict__ perm,
                            const float* __restrict__ scores,
                            float* __restrict__ xout,
                            unsigned int k, int F) {
    const unsigned int r = blockIdx.x * blockDim.x + threadIdx.x;
    if (r >= k) return;
    const unsigned int node = perm[r];
    const float g = tanhf(scores[node]);
    for (int f = 0; f < F; ++f)
        xout[(size_t)r * F + f] = x[(size_t)node * F + f] * g;
}

// ---------------------------------------------------------------------------
// Stage 4: edge remap/filter.  Dominant stream: 256 MB in, ~192 MB out.
// NT loads/stores keep the 4 MB remap table hot in L2; prefetch hides latency.
// ---------------------------------------------------------------------------
__global__ void edge_kernel(const long long* __restrict__ ei, long long E,
                            const int* __restrict__ nidx,
                            float* __restrict__ out_edge,   // 2E floats
                            float* __restrict__ out_mask) { // E floats
    long long i = (long long)blockIdx.x * blockDim.x + threadIdx.x;
    const long long stride = (long long)gridDim.x * blockDim.x;
    for (; i < E; i += stride) {
        __builtin_prefetch(&ei[i + stride], 0, 1);
        __builtin_prefetch(&ei[E + i + stride], 0, 1);
        const long long s = __builtin_nontemporal_load(&ei[i]);
        const long long d = __builtin_nontemporal_load(&ei[E + i]);
        const int ns = nidx[(int)s];
        const int nd = nidx[(int)d];
        __builtin_nontemporal_store((float)ns, &out_edge[i]);
        __builtin_nontemporal_store((float)nd, &out_edge[E + i]);
        __builtin_nontemporal_store((ns >= 0 && nd >= 0) ? 1.0f : 0.0f, &out_mask[i]);
    }
}

extern "C" void kernel_launch(void* const* d_in, const int* in_sizes, int n_in,
                              void* d_out, int out_size, void* d_ws, size_t ws_size,
                              hipStream_t stream) {
    (void)n_in; (void)out_size; (void)ws_size;
    const float*     x  = (const float*)d_in[0];
    const long long* ei = (const long long*)d_in[1];
    const float*     w  = (const float*)d_in[2];

    const int       F = in_sizes[2];
    const int       N = in_sizes[0] / F;
    const long long E = (long long)in_sizes[1] / 2;
    const unsigned int k = (unsigned int)((N + 1) / 2);   // ceil(N/2)

    // Workspace layout (~14 MB for N=1M, k=500k)
    unsigned int* keys   = (unsigned int*)d_ws;           // N u32
    float*        scores = (float*)(keys + N);            // N f32
    int*          nidx   = (int*)(scores + N);            // N i32
    unsigned int* perm   = (unsigned int*)(nidx + N);     // k u32
    unsigned int* hist4  = perm + k;                      // 4*256 u32
    unsigned int* state  = hist4 + 1024;                  // 8 u32

    // Output layout: [x_out k*F][new_edge_index 2E][edge_mask E] (floats)
    float* out      = (float*)d_out;
    float* xout     = out;
    float* out_edge = out + (size_t)k * F;
    float* out_mask = out_edge + 2 * (size_t)E;

    init_kernel<<<4, 256, 0, stream>>>(hist4, state, k);

    const int waves    = (N + 15) / 16;
    const int blocks_s = (waves * 32 + 255) / 256;
    score_wmma_kernel<<<blocks_s, 256, 0, stream>>>(x, w, N, F, keys, scores);

    const int hb = 1024;
    for (int p = 0; p < 4; ++p) {
        hist_kernel<<<hb, 256, 0, stream>>>(keys, N, state, hist4 + 256 * p, p);
        scan_kernel<<<1, 32, 0, stream>>>(hist4 + 256 * p, state, p);
    }
    compact_kernel<<<hb, 256, 0, stream>>>(keys, N, state, nidx, perm, k);

    xout_kernel<<<(int)((k + 255) / 256), 256, 0, stream>>>(x, perm, scores, xout, k, F);

    long long eb = (E + 1023) / 1024;                     // 4 edges/thread
    if (eb < 1) eb = 1;
    edge_kernel<<<(int)eb, 256, 0, stream>>>(ei, E, nidx, out_edge, out_mask);
}
